// SwinTransformerBlock_64149631533663
// MI455X (gfx1250) — compile-verified, matrix operand order checked by reading, not run on hardware
//
#include <hip/hip_runtime.h>

// ---------------------------------------------------------------------------
// Types
// ---------------------------------------------------------------------------
typedef __bf16 bf16_t;
typedef bf16_t v16bf __attribute__((ext_vector_type(16)));
typedef bf16_t v8bf  __attribute__((ext_vector_type(8)));
typedef float  v8f   __attribute__((ext_vector_type(8)));
typedef unsigned v4u __attribute__((ext_vector_type(4)));
typedef int      v4i __attribute__((ext_vector_type(4)));
typedef int      v8i __attribute__((ext_vector_type(8)));

// ---------------------------------------------------------------------------
// Problem constants (B=64, H=W=56, C=192, WS=7, SHIFT=3, NH=6)
// ---------------------------------------------------------------------------
#define C_DIM   192
#define NHEAD   6
#define HDIM    32
#define WSZ     7
#define NW      49
#define SHF     3
#define HW      56
#define LTOK    3136
#define MTOT    200704      // B*L == B_*49
#define HIDD    768
#define QKVN    576

// ---------------------------------------------------------------------------
// CDNA5 data-movement feature detection (device pass only)
// ---------------------------------------------------------------------------
#if defined(__HIP_DEVICE_COMPILE__) && \
    __has_builtin(__builtin_amdgcn_tensor_load_to_lds) && \
    __has_builtin(__builtin_amdgcn_s_wait_tensorcnt)
#define USE_TDM 1
#elif defined(__HIP_DEVICE_COMPILE__) && \
    __has_builtin(__builtin_amdgcn_global_load_async_to_lds_b128)
#define USE_ASYNC 1
#endif

static __device__ __forceinline__ bf16_t to_bf16(float f) {
    unsigned u = __builtin_bit_cast(unsigned, f);
    unsigned r = (u + 0x7FFFu + ((u >> 16) & 1u)) >> 16;
    return __builtin_bit_cast(bf16_t, (unsigned short)r);
}

static __device__ __forceinline__ v8f wmma_bf16(v16bf a, v16bf b, v8f c) {
    return __builtin_amdgcn_wmma_f32_16x16x32_bf16(
        false, a, false, b, (short)0, c, false, false);
}

static __device__ __forceinline__ v16bf ld_frag(const bf16_t* p0, const bf16_t* p1) {
    v8bf a = *(const v8bf*)p0;
    v8bf b = *(const v8bf*)p1;
    v16bf r;
#pragma unroll
    for (int i = 0; i < 8; ++i) { r[i] = a[i]; r[i + 8] = b[i]; }
    return r;
}

static __device__ __forceinline__ v16bf zero_frag() {
    v16bf r;
#pragma unroll
    for (int i = 0; i < 16; ++i) r[i] = to_bf16(0.0f);
    return r;
}

static __device__ __forceinline__ unsigned lds_off_bytes(void* p) {
    // generic -> LDS addrspace(3) -> 32-bit LDS byte offset
    return (unsigned)(unsigned long long)(__attribute__((address_space(3))) void*)p;
}

#if defined(USE_TDM)
// Issue one TDM DMA: 64 rows x 32 bf16 (= 16 dwords/row) tile, row stride
// strideElems bf16, from global gsrc into LDS at ldsOff. 2D descriptor,
// data_size=4B, count=1, type=2 ("image"). Tracked by TENSORcnt.
static __device__ __forceinline__ void tdm_tile_64x32(const bf16_t* gsrc,
                                                      int strideElems,
                                                      unsigned ldsOff) {
    unsigned long long ga = (unsigned long long)gsrc;
    v4u g0 = { 1u,                                    // count=1, user mode
               ldsOff,                                // lds_addr (bytes)
               (unsigned)ga,                          // global_addr[31:0]
               ((unsigned)(ga >> 32) & 0x01FFFFFFu)   // global_addr[56:32]
                   | (2u << 30) };                    // type=2
    v8i g1 = { (int)(2u << 16),     // wg_mask=0 | data_size=2 (4B)
               (int)(16u << 16),    // tensor_dim0[15:0]=16 dw @bits63:48
               (int)(64u << 16),    // dim0[31:16]=0 | tensor_dim1[15:0]=64
               (int)(16u << 16),    // dim1[31:16]=0 | tile_dim0=16 dw
               (int)64,             // tile_dim1=64 | tile_dim2=0
               (int)(strideElems >> 1),   // tensor_dim0_stride lo32 (dwords)
               0, 0 };              // stride hi | dim1_stride (unused, 2D)
    v4i z4 = { 0, 0, 0, 0 };
#if defined(__clang_major__) && __clang_major__ >= 23
    v8i z8 = { 0, 0, 0, 0, 0, 0, 0, 0 };
    __builtin_amdgcn_tensor_load_to_lds(g0, g1, z4, z4, z8, 0);
#else
    __builtin_amdgcn_tensor_load_to_lds(g0, g1, z4, z4, 0);
#endif
}
#endif

#if defined(USE_ASYNC)
static __device__ __forceinline__ void async_cp_b128(const bf16_t* g, bf16_t* l) {
    __builtin_amdgcn_global_load_async_to_lds_b128(
        (__attribute__((address_space(1))) void*)(unsigned long long)g,
        (__attribute__((address_space(3))) void*)l, 0, 0);
}
static __device__ __forceinline__ void wait_async(int more) {
#if __has_builtin(__builtin_amdgcn_s_wait_asynccnt)
    if (more) __builtin_amdgcn_s_wait_asynccnt(2);
    else      __builtin_amdgcn_s_wait_asynccnt(0);
#else
    if (more) asm volatile("s_wait_asynccnt 0x2" ::: "memory");
    else      asm volatile("s_wait_asynccnt 0x0" ::: "memory");
#endif
}
#endif

// ---------------------------------------------------------------------------
// Weight transpose + f32->bf16 convert:  dst[n*K + k] = src[k*N + n]
// ---------------------------------------------------------------------------
__global__ void convert_wt_kernel(const float* __restrict__ src,
                                  bf16_t* __restrict__ dst, int K, int N) {
    long t = (long)blockIdx.x * 256 + threadIdx.x;
    if (t >= (long)K * N) return;
    int n = (int)(t / K);
    int k = (int)(t % K);
    dst[t] = to_bf16(src[(long)k * N + n]);
}

// ---------------------------------------------------------------------------
// LayerNorm1 + cyclic shift + window partition -> bf16 [B_*49, 192]
// ---------------------------------------------------------------------------
__global__ __launch_bounds__(256) void ln1_shift_window_kernel(
    const float* __restrict__ x, const float* __restrict__ g,
    const float* __restrict__ b, bf16_t* __restrict__ out) {
    int row  = blockIdx.x * 8 + (threadIdx.x >> 5);
    int lane = threadIdx.x & 31;
    if (row >= MTOT) return;

    int wi = row / NW, nidx = row % NW;
    int bb = wi >> 6, widx = wi & 63;
    int wh = widx >> 3, ww = widx & 7;
    int i = nidx / WSZ, j = nidx % WSZ;
    int h  = (wh * WSZ + i + SHF) % HW;
    int wc = (ww * WSZ + j + SHF) % HW;
    long src = ((long)bb * LTOK + h * HW + wc) * C_DIM;

    float vals[6];
    float s = 0.0f;
#pragma unroll
    for (int t = 0; t < 6; ++t) { vals[t] = x[src + t * 32 + lane]; s += vals[t]; }
#pragma unroll
    for (int o = 16; o >= 1; o >>= 1) s += __shfl_xor(s, o, 32);
    float mu = s * (1.0f / 192.0f);
    float vs = 0.0f;
#pragma unroll
    for (int t = 0; t < 6; ++t) { float d = vals[t] - mu; vs += d * d; }
#pragma unroll
    for (int o = 16; o >= 1; o >>= 1) vs += __shfl_xor(vs, o, 32);
    float rs = rsqrtf(vs * (1.0f / 192.0f) + 1e-5f);
#pragma unroll
    for (int t = 0; t < 6; ++t) {
        int c = t * 32 + lane;
        out[(long)row * C_DIM + c] = to_bf16((vals[t] - mu) * rs * g[c] + b[c]);
    }
}

// ---------------------------------------------------------------------------
// LayerNorm2 (identity mapping): f32 in -> bf16 out
// ---------------------------------------------------------------------------
__global__ __launch_bounds__(256) void ln2_kernel(
    const float* __restrict__ x, const float* __restrict__ g,
    const float* __restrict__ b, bf16_t* __restrict__ out) {
    int row  = blockIdx.x * 8 + (threadIdx.x >> 5);
    int lane = threadIdx.x & 31;
    if (row >= MTOT) return;
    long src = (long)row * C_DIM;
    float vals[6];
    float s = 0.0f;
#pragma unroll
    for (int t = 0; t < 6; ++t) { vals[t] = x[src + t * 32 + lane]; s += vals[t]; }
#pragma unroll
    for (int o = 16; o >= 1; o >>= 1) s += __shfl_xor(s, o, 32);
    float mu = s * (1.0f / 192.0f);
    float vs = 0.0f;
#pragma unroll
    for (int t = 0; t < 6; ++t) { float d = vals[t] - mu; vs += d * d; }
#pragma unroll
    for (int o = 16; o >= 1; o >>= 1) vs += __shfl_xor(vs, o, 32);
    float rs = rsqrtf(vs * (1.0f / 192.0f) + 1e-5f);
#pragma unroll
    for (int t = 0; t < 6; ++t) {
        int c = t * 32 + lane;
        out[src + c] = to_bf16((vals[t] - mu) * rs * g[c] + b[c]);
    }
}

// ---------------------------------------------------------------------------
// Generic bf16 WMMA GEMM:  out[M,N] = A[M,K] * Bt[N,K]^T (+bias, epilogue)
//   BM=64 BN=64 BK=32, 256 threads = 8 waves (2x4), wave tile 32x16
//   Double-buffered LDS; tiles staged by TDM / async-to-LDS / sync copy.
// epi: 0 = +bias -> bf16                               (QKV)
//      1 = +bias, window-reverse+roll, +resid -> f32   (PROJ)
//      2 = gelu(+bias) -> bf16                         (FC1)
//      3 = +bias +resid -> f32 in place                (FC2)
// ---------------------------------------------------------------------------
__global__ __launch_bounds__(256) void gemm_bf16_kernel(
    const bf16_t* __restrict__ A, const bf16_t* __restrict__ Bt,
    const float* __restrict__ bias, const float* addsrc,
    float* outf, bf16_t* __restrict__ outb, int K, int N, int epi) {
    __shared__ __attribute__((aligned(16))) bf16_t sA[2][64 * 32];
    __shared__ __attribute__((aligned(16))) bf16_t sB[2][64 * 32];

    const int m0 = blockIdx.x * 64;
    const int n0 = blockIdx.y * 64;
    const int tid = threadIdx.x;
    const int w = tid >> 5, l = tid & 31;
    const int wm = w >> 2, wn = w & 3;
    const int half = l >> 4, r = l & 15;
    const int ldrow = tid >> 2, ldcol = (tid & 3) * 8;

    v8f acc0 = {}, acc1 = {};

#if defined(USE_TDM) || defined(USE_ASYNC)
    // ---------------- double-buffered async pipeline ----------------
#if defined(USE_TDM)
    if (tid == 0) {
        tdm_tile_64x32(A + (long)m0 * K, K, lds_off_bytes(&sA[0][0]));
        tdm_tile_64x32(Bt + (long)n0 * K, K, lds_off_bytes(&sB[0][0]));
    }
#else
    async_cp_b128(&A[(long)(m0 + ldrow) * K + ldcol], &sA[0][ldrow * 32 + ldcol]);
    async_cp_b128(&Bt[(long)(n0 + ldrow) * K + ldcol], &sB[0][ldrow * 32 + ldcol]);
#endif
    int cur = 0;
    for (int kk = 0; kk < K; kk += 32) {
        const int more = (kk + 32 < K);
        __syncthreads();            // everyone done reading buffer cur^1
        if (more) {
#if defined(USE_TDM)
            if (tid == 0) {
                tdm_tile_64x32(A + (long)m0 * K + kk + 32, K,
                               lds_off_bytes(&sA[cur ^ 1][0]));
                tdm_tile_64x32(Bt + (long)n0 * K + kk + 32, K,
                               lds_off_bytes(&sB[cur ^ 1][0]));
            }
#else
            async_cp_b128(&A[(long)(m0 + ldrow) * K + kk + 32 + ldcol],
                          &sA[cur ^ 1][ldrow * 32 + ldcol]);
            async_cp_b128(&Bt[(long)(n0 + ldrow) * K + kk + 32 + ldcol],
                          &sB[cur ^ 1][ldrow * 32 + ldcol]);
#endif
        }
#if defined(USE_TDM)
        if (tid == 0) {
            if (more) __builtin_amdgcn_s_wait_tensorcnt(2);
            else      __builtin_amdgcn_s_wait_tensorcnt(0);
        }
#else
        wait_async(more);
#endif
        __syncthreads();            // tile `cur` landed in LDS for all waves

        const bf16_t* pb  = &sB[cur][(wn * 16 + r) * 32];
        v16bf fb  = ld_frag(pb + 8 * half, pb + 16 + 8 * half);
        const bf16_t* pa0 = &sA[cur][(wm * 32 + r) * 32];
        const bf16_t* pa1 = &sA[cur][(wm * 32 + 16 + r) * 32];
        v16bf fa0 = ld_frag(pa0 + 8 * half, pa0 + 16 + 8 * half);
        v16bf fa1 = ld_frag(pa1 + 8 * half, pa1 + 16 + 8 * half);
        acc0 = wmma_bf16(fa0, fb, acc0);
        acc1 = wmma_bf16(fa1, fb, acc1);
        cur ^= 1;
    }
#else
    // ---------------- synchronous fallback ----------------
    for (int kk = 0; kk < K; kk += 32) {
        *(v8bf*)&sA[0][ldrow * 32 + ldcol] =
            *(const v8bf*)&A[(long)(m0 + ldrow) * K + kk + ldcol];
        *(v8bf*)&sB[0][ldrow * 32 + ldcol] =
            *(const v8bf*)&Bt[(long)(n0 + ldrow) * K + kk + ldcol];
        if (kk + 32 < K) {
            __builtin_prefetch(&A[(long)(m0 + ldrow) * K + kk + 32 + ldcol], 0, 1);
            __builtin_prefetch(&Bt[(long)(n0 + ldrow) * K + kk + 32 + ldcol], 0, 1);
        }
        __syncthreads();
        const bf16_t* pb  = &sB[0][(wn * 16 + r) * 32];
        v16bf fb  = ld_frag(pb + 8 * half, pb + 16 + 8 * half);
        const bf16_t* pa0 = &sA[0][(wm * 32 + r) * 32];
        const bf16_t* pa1 = &sA[0][(wm * 32 + 16 + r) * 32];
        v16bf fa0 = ld_frag(pa0 + 8 * half, pa0 + 16 + 8 * half);
        v16bf fa1 = ld_frag(pa1 + 8 * half, pa1 + 16 + 8 * half);
        acc0 = wmma_bf16(fa0, fb, acc0);
        acc1 = wmma_bf16(fa1, fb, acc1);
        __syncthreads();
    }
#endif

    const int col = n0 + wn * 16 + r;
    const float bv = bias[col];
#pragma unroll
    for (int s = 0; s < 2; ++s) {
        v8f acc = s ? acc1 : acc0;
#pragma unroll
        for (int v = 0; v < 8; ++v) {
            int row = m0 + wm * 32 + s * 16 + half * 8 + v;
            float val = acc[v] + bv;
            if (epi == 0) {
                outb[(long)row * N + col] = to_bf16(val);
            } else if (epi == 2) {
                float g = 0.5f * val * (1.0f + erff(val * 0.70710678118654752f));
                outb[(long)row * N + col] = to_bf16(g);
            } else if (epi == 3) {
                long o = (long)row * C_DIM + col;
                outf[o] = addsrc[o] + val;
            } else { // epi == 1
                int wi = row / NW, nidx = row % NW;
                int bb = wi >> 6, widx = wi & 63;
                int wh = widx >> 3, ww = widx & 7;
                int i = nidx / WSZ, j = nidx % WSZ;
                int h  = (wh * WSZ + i + SHF) % HW;
                int wc = (ww * WSZ + j + SHF) % HW;
                long o = ((long)bb * LTOK + h * HW + wc) * C_DIM + col;
                outf[o] = addsrc[o] + val;
            }
        }
    }
}

// ---------------------------------------------------------------------------
// Windowed attention: one block (128 thr = 4 waves) per (window, head).
// ---------------------------------------------------------------------------
__global__ __launch_bounds__(128) void attn_kernel(
    const bf16_t* __restrict__ qkv, const float* __restrict__ rpb,
    bf16_t* __restrict__ out) {
    __shared__ float srpb[169];
    __shared__ int sg[NW];
    __shared__ __attribute__((aligned(16))) bf16_t sVt[32 * 64];
    __shared__ __attribute__((aligned(16))) bf16_t sP[4][16 * 64];

    const int bx = blockIdx.x;
    const int wi = bx / NHEAD, head = bx % NHEAD;
    const int widx = wi & 63, wh = widx >> 3, ww = widx & 7;
    const int tid = threadIdx.x;
    const int wave = tid >> 5, l = tid & 31, half = l >> 4, r = l & 15;

    for (int i = tid; i < 169; i += 128) srpb[i] = rpb[i * NHEAD + head];
    if (tid < NW) {
        int i = tid / WSZ, j = tid % WSZ;
        int hh = wh * WSZ + i, wwp = ww * WSZ + j;
        int rh = (hh < 49) ? 0 : ((hh < 53) ? 1 : 2);
        int rw = (wwp < 49) ? 0 : ((wwp < 53) ? 1 : 2);
        sg[tid] = rh * 3 + rw;
    }
    for (int idx = tid; idx < 2048; idx += 128) {
        int kk = idx >> 5, c = idx & 31;
        bf16_t v = (kk < NW)
            ? qkv[(long)(wi * NW + kk) * QKVN + 384 + head * HDIM + c]
            : to_bf16(0.0f);
        sVt[c * 64 + kk] = v;
    }
    __syncthreads();

    const int mt = wave;

    v16bf fq;
    {
        int qrow = mt * 16 + r;
        if (qrow < NW) {
            const bf16_t* p = qkv + (long)(wi * NW + qrow) * QKVN + head * HDIM;
            fq = ld_frag(p + 8 * half, p + 16 + 8 * half);
        } else fq = zero_frag();
    }

    v8f sacc[4];
#pragma unroll
    for (int nt = 0; nt < 4; ++nt) {
        v16bf fk;
        int krow = nt * 16 + r;
        if (krow < NW) {
            const bf16_t* p = qkv + (long)(wi * NW + krow) * QKVN + C_DIM + head * HDIM;
            fk = ld_frag(p + 8 * half, p + 16 + 8 * half);
        } else fk = zero_frag();
        v8f z = {};
        sacc[nt] = wmma_bf16(fq, fk, z);
    }

    const float scale = 0.17677669529663687f;   // 1/sqrt(32)
#pragma unroll
    for (int v = 0; v < 8; ++v) {
        int mloc = half * 8 + v;
        int m = mt * 16 + mloc;
        int mq = (m < NW) ? m : (NW - 1);
        int iq = mq / WSZ, jq = mq % WSZ, gq = sg[mq];
        float sv[4];
        float mx = -3.0e38f;
#pragma unroll
        for (int nt = 0; nt < 4; ++nt) {
            int nk = nt * 16 + r;
            float val;
            if (nk < NW) {
                int ik = nk / WSZ, jk = nk % WSZ;
                int idx = (iq - ik + 6) * 13 + (jq - jk + 6);
                float mkv = (gq == sg[nk]) ? 0.0f : -100.0f;
                val = sacc[nt][v] * scale + srpb[idx] + mkv;
            } else val = -3.0e38f;
            sv[nt] = val;
            mx = fmaxf(mx, val);
        }
#pragma unroll
        for (int o = 1; o < 16; o <<= 1) mx = fmaxf(mx, __shfl_xor(mx, o, 32));
        float sum = 0.0f;
        float pv[4];
#pragma unroll
        for (int nt = 0; nt < 4; ++nt) {
            float p = (nt * 16 + r < NW) ? __expf(sv[nt] - mx) : 0.0f;
            pv[nt] = p;
            sum += p;
        }
#pragma unroll
        for (int o = 1; o < 16; o <<= 1) sum += __shfl_xor(sum, o, 32);
        float inv = 1.0f / sum;
#pragma unroll
        for (int nt = 0; nt < 4; ++nt)
            sP[mt][mloc * 64 + nt * 16 + r] = to_bf16(pv[nt] * inv);
    }
    __syncthreads();

    v8f oacc0 = {}, oacc1 = {};
#pragma unroll
    for (int kc = 0; kc < 2; ++kc) {
        const bf16_t* pp = &sP[mt][r * 64 + kc * 32];
        v16bf fp = ld_frag(pp + 8 * half, pp + 16 + 8 * half);
        const bf16_t* pv0 = &sVt[(0 * 16 + r) * 64 + kc * 32];
        const bf16_t* pv1 = &sVt[(1 * 16 + r) * 64 + kc * 32];
        v16bf fv0 = ld_frag(pv0 + 8 * half, pv0 + 16 + 8 * half);
        v16bf fv1 = ld_frag(pv1 + 8 * half, pv1 + 16 + 8 * half);
        oacc0 = wmma_bf16(fp, fv0, oacc0);
        oacc1 = wmma_bf16(fp, fv1, oacc1);
    }

#pragma unroll
    for (int tn = 0; tn < 2; ++tn) {
        v8f oa = tn ? oacc1 : oacc0;
#pragma unroll
        for (int v = 0; v < 8; ++v) {
            int m = mt * 16 + half * 8 + v;
            if (m < NW) {
                int col = tn * 16 + r;
                out[(long)(wi * NW + m) * C_DIM + head * HDIM + col] = to_bf16(oa[v]);
            }
        }
    }
}

// ---------------------------------------------------------------------------
// Host-side launch
// ---------------------------------------------------------------------------
extern "C" void kernel_launch(void* const* d_in, const int* in_sizes, int n_in,
                              void* d_out, int out_size, void* d_ws, size_t ws_size,
                              hipStream_t stream) {
    const float* x      = (const float*)d_in[0];
    const float* n1g    = (const float*)d_in[1];
    const float* n1b    = (const float*)d_in[2];
    const float* qkv_w  = (const float*)d_in[3];
    const float* qkv_b  = (const float*)d_in[4];
    const float* rpb    = (const float*)d_in[5];
    const float* proj_w = (const float*)d_in[6];
    const float* proj_b = (const float*)d_in[7];
    const float* n2g    = (const float*)d_in[8];
    const float* n2b    = (const float*)d_in[9];
    const float* fc1_w  = (const float*)d_in[10];
    const float* fc1_b  = (const float*)d_in[11];
    const float* fc2_w  = (const float*)d_in[12];
    const float* fc2_b  = (const float*)d_in[13];
    float* out = (float*)d_out;
    char* ws = (char*)d_ws;

    const size_t OFF_WQKV = 0;
    const size_t OFF_WPRJ = OFF_WQKV + (size_t)QKVN * C_DIM * 2;
    const size_t OFF_WFC1 = OFF_WPRJ + (size_t)C_DIM * C_DIM * 2;
    const size_t OFF_WFC2 = OFF_WFC1 + (size_t)HIDD * C_DIM * 2;
    const size_t OFF_A    = OFF_WFC2 + (size_t)C_DIM * HIDD * 2;
    const size_t OFF_BC   = OFF_A + (size_t)MTOT * C_DIM * 2;

    bf16_t* wqkv_t = (bf16_t*)(ws + OFF_WQKV);
    bf16_t* wprj_t = (bf16_t*)(ws + OFF_WPRJ);
    bf16_t* wfc1_t = (bf16_t*)(ws + OFF_WFC1);
    bf16_t* wfc2_t = (bf16_t*)(ws + OFF_WFC2);
    bf16_t* bufA   = (bf16_t*)(ws + OFF_A);    // hwin -> attn_out -> ln2
    bf16_t* bufBC  = (bf16_t*)(ws + OFF_BC);   // qkv -> gelu

    convert_wt_kernel<<<(C_DIM * QKVN + 255) / 256, 256, 0, stream>>>(qkv_w, wqkv_t, C_DIM, QKVN);
    convert_wt_kernel<<<(C_DIM * C_DIM + 255) / 256, 256, 0, stream>>>(proj_w, wprj_t, C_DIM, C_DIM);
    convert_wt_kernel<<<(C_DIM * HIDD + 255) / 256, 256, 0, stream>>>(fc1_w, wfc1_t, C_DIM, HIDD);
    convert_wt_kernel<<<(HIDD * C_DIM + 255) / 256, 256, 0, stream>>>(fc2_w, wfc2_t, HIDD, C_DIM);

    ln1_shift_window_kernel<<<MTOT / 8, 256, 0, stream>>>(x, n1g, n1b, bufA);

    gemm_bf16_kernel<<<dim3(MTOT / 64, QKVN / 64), 256, 0, stream>>>(
        bufA, wqkv_t, qkv_b, nullptr, nullptr, bufBC, C_DIM, QKVN, 0);

    attn_kernel<<<(MTOT / NW) * NHEAD, 128, 0, stream>>>(bufBC, rpb, bufA);

    gemm_bf16_kernel<<<dim3(MTOT / 64, C_DIM / 64), 256, 0, stream>>>(
        bufA, wprj_t, proj_b, x, out, nullptr, C_DIM, C_DIM, 1);

    ln2_kernel<<<MTOT / 8, 256, 0, stream>>>(out, n2g, n2b, bufA);

    gemm_bf16_kernel<<<dim3(MTOT / 64, HIDD / 64), 256, 0, stream>>>(
        bufA, wfc1_t, fc1_b, nullptr, nullptr, bufBC, C_DIM, HIDD, 2);

    gemm_bf16_kernel<<<dim3(MTOT / 64, C_DIM / 64), 256, 0, stream>>>(
        bufBC, wfc2_t, fc2_b, out, out, nullptr, HIDD, C_DIM, 3);
}
